// InteractionBlock_22686017258127
// MI455X (gfx1250) — compile-verified, matrix operand order checked by reading, not run on hardware
//
#include <hip/hip_runtime.h>
#include <hip/hip_bf16.h>
#include <stdint.h>

typedef __attribute__((ext_vector_type(16))) __bf16 v16bf;
typedef __attribute__((ext_vector_type(2)))  __bf16 v2bf;
typedef __attribute__((ext_vector_type(8)))  float  v8f;
typedef __attribute__((ext_vector_type(4)))  unsigned int v4u;
typedef __attribute__((ext_vector_type(8)))  int v8i;
typedef __attribute__((ext_vector_type(4)))  int v4i;

#define LN2F 0.69314718056f
#define PIF  3.14159265f
#define F_DIM 64
#define R_DIM 300
#define KP1   320       // R padded to multiple of 32
#define KT1   10        // K-tiles in layer 1
#define EDGE_WAVES 4    // waves per block in edge kernel

#if __has_builtin(__builtin_amdgcn_tensor_load_to_lds)
#define HAVE_TDM 1
#else
#define HAVE_TDM 0
#endif

__device__ __forceinline__ float sspf(float x) {
    // shifted softplus, numerically stable
    return fmaxf(x, 0.f) + __logf(1.f + __expf(-fabsf(x))) - LN2F;
}

// K offset of bf16 pair slot p for this half-wave (16-bit A/B fragment layout)
__device__ __forceinline__ int kpair(int p, int half) {
    return ((p >> 2) << 4) + ((p & 3) << 1) + (half << 3);
}

__device__ __forceinline__ v8f wmma_bf16(v16bf a, v16bf b, v8f c) {
    return __builtin_amdgcn_wmma_f32_16x16x32_bf16(false, a, false, b,
                                                   (short)0, c, false, false);
}

// A fragment from global fp32 row-major [*, ldk], row fixed per lane, K guarded
__device__ __forceinline__ v16bf load_a_global(const float* __restrict__ src,
                                               int row, int ldk, int k0,
                                               int kmax, int half) {
    v16bf a;
    const float* rp = src + (size_t)row * ldk;
#pragma unroll
    for (int p = 0; p < 8; ++p) {
        int k = k0 + kpair(p, half);
        float x0 = (k     < kmax) ? rp[k]     : 0.f;
        float x1 = (k + 1 < kmax) ? rp[k + 1] : 0.f;
        a[2 * p]     = (__bf16)x0;
        a[2 * p + 1] = (__bf16)x1;
    }
    return a;
}

// A fragment from an LDS fp32 row (32 floats per row, one K-tile)
__device__ __forceinline__ v16bf load_a_lds_f32(const float* rp, int half) {
    v16bf a;
#pragma unroll
    for (int p = 0; p < 8; ++p) {
        int k = kpair(p, half);
        float x0 = rp[k];
        float x1 = rp[k + 1];
        a[2 * p]     = (__bf16)x0;
        a[2 * p + 1] = (__bf16)x1;
    }
    return a;
}

// A or B fragment from LDS bf16 where the per-lane K-run is contiguous:
// layout [row][ldk] bf16; for B fragments "row" is the output column n.
__device__ __forceinline__ v16bf load_frag_lds(const __bf16* s, int ldk,
                                               int row, int k0, int half) {
    v16bf f;
    const __bf16* rp = s + row * ldk;
#pragma unroll
    for (int p = 0; p < 8; ++p) {
        int k = k0 + kpair(p, half);
        v2bf t = *(const v2bf*)&rp[k];   // ds_load_b32 (pair of bf16)
        f[2 * p]     = t[0];
        f[2 * p + 1] = t[1];
    }
    return f;
}

// Stage fp32 weights [K][64] as transposed bf16 [64][KP] in LDS (zero-pad K)
__device__ __forceinline__ void fill_bt(__bf16* bt, const float* __restrict__ fw,
                                        int kmax, int KP, int tid, int nthr) {
    for (int idx = tid; idx < F_DIM * KP; idx += nthr) {
        int n = idx / KP, k = idx - n * KP;
        bt[idx] = (__bf16)((k < kmax) ? fw[k * F_DIM + n] : 0.f);
    }
}

#if HAVE_TDM
// Issue a 2D TDM tile load: tile_d0 x tile_d1 elements of 4B each from
// global (row stride stride0 elements) into LDS at lds_addr. OOB columns
// (x >= tensor_d0) are zero-filled by the TDM.
__device__ __forceinline__ void tdm_load_2d(uint32_t lds_addr, const void* gaddr,
                                            uint32_t tensor_d0, uint32_t tensor_d1,
                                            uint32_t tile_d0, uint32_t tile_d1,
                                            uint32_t stride0) {
    uint64_t ga = (uint64_t)(uintptr_t)gaddr;
    v4u g0;
    g0[0] = 1u;                                   // count=1 (valid), user mode
    g0[1] = lds_addr;                             // lds_addr [63:32]
    g0[2] = (uint32_t)(ga & 0xFFFFFFFFu);         // global_addr[31:0]
    g0[3] = (uint32_t)((ga >> 32) & 0x01FFFFFFu)  // global_addr[56:32]
            | (2u << 30);                         // type=2 ("image")
    v8i g1;
    g1[0] = (int)(2u << 16);                      // data_size=2 -> 4 bytes
    g1[1] = (int)((tensor_d0 & 0xFFFFu) << 16);   // tensor_dim0[15:0]
    g1[2] = (int)(((tensor_d0 >> 16) & 0xFFFFu) | ((tensor_d1 & 0xFFFFu) << 16));
    g1[3] = (int)(((tensor_d1 >> 16) & 0xFFFFu) | ((tile_d0 & 0xFFFFu) << 16));
    g1[4] = (int)(tile_d1 & 0xFFFFu);             // tile_dim1; tile_dim2=0
    g1[5] = (int)stride0;                         // tensor_dim0_stride[31:0]
    g1[6] = 0;                                    // stride0[47:32], dim1_stride lo
    g1[7] = 0;                                    // dim1_stride hi
    v4i z4 = {0, 0, 0, 0};
#if __clang_major__ >= 23
    v8i z8 = {0, 0, 0, 0, 0, 0, 0, 0};
    __builtin_amdgcn_tensor_load_to_lds(g0, g1, z4, z4, z8, 0);
#else
    __builtin_amdgcn_tensor_load_to_lds(g0, g1, z4, z4, 0);
#endif
}
#endif

__global__ void zero_kernel(float* __restrict__ p, int n) {
    int i = blockIdx.x * blockDim.x + threadIdx.x;
    if (i < n) p[i] = 0.f;
}

// ---------------- h = x @ l1w + l1b ----------------
__global__ __launch_bounds__(256) void node_linear1(
    const float* __restrict__ x, const float* __restrict__ w,
    const float* __restrict__ b, float* __restrict__ out, int ntiles) {
    __shared__ __bf16 bt[F_DIM * F_DIM];
    fill_bt(bt, w, F_DIM, F_DIM, threadIdx.x, 256);
    __syncthreads();

    const int lane = threadIdx.x & 31, wave = threadIdx.x >> 5;
    const int half = lane >> 4, l15 = lane & 15;
    v8f z = {0.f, 0.f, 0.f, 0.f, 0.f, 0.f, 0.f, 0.f};

    for (int tile = blockIdx.x * 8 + wave; tile < ntiles; tile += gridDim.x * 8) {
        const int m0 = tile * 16;
        v8f acc[4] = {z, z, z, z};
#pragma unroll
        for (int k0 = 0; k0 < F_DIM; k0 += 32) {
            v16bf a = load_a_global(x, m0 + l15, F_DIM, k0, F_DIM, half);
#pragma unroll
            for (int nt = 0; nt < 4; ++nt) {
                v16bf bb = load_frag_lds(bt, F_DIM, nt * 16 + l15, k0, half);
                acc[nt] = wmma_bf16(a, bb, acc[nt]);
            }
        }
#pragma unroll
        for (int nt = 0; nt < 4; ++nt) {
            int c = nt * 16 + l15;
            float bias = b[c];
#pragma unroll
            for (int r = 0; r < 8; ++r)
                out[(size_t)(m0 + r + 8 * half) * F_DIM + c] = acc[nt][r] + bias;
        }
    }
}

// -------- fused filter-net + cutoff + gather + scatter-add --------
// Each wave: one 32-edge x 64-feature tile; rbf tiles staged into LDS by the
// Tensor Data Mover (double buffered on TENSORcnt), weights cached in LDS bf16.
__global__ __launch_bounds__(128) void edge_kernel(
    const long long* __restrict__ edge_index,  // [2][E] int64
    const float* __restrict__ rbf,             // [E][300]
    const float* __restrict__ dist,            // [E]
    const float* __restrict__ cutoffp,         // scalar
    const float* __restrict__ fw1, const float* __restrict__ fb1,
    const float* __restrict__ fw2, const float* __restrict__ fb2,
    const float* __restrict__ h,               // [N][64]
    float* __restrict__ agg,                   // [N][64]
    int E, int ntiles32) {
    __shared__ __bf16 bt1[F_DIM * KP1];              // fw1^T bf16, 40KB
    __shared__ __bf16 bt2[F_DIM * F_DIM];            // fw2^T bf16,  8KB
    __shared__ float  rbuf[EDGE_WAVES][2][32 * 32];  // TDM tiles,  32KB
    __shared__ __bf16 sc[EDGE_WAVES][32 * F_DIM];    // relayout,   16KB
    __shared__ float  wfac[EDGE_WAVES][32];
    __shared__ int    ssrc[EDGE_WAVES][32], sdst[EDGE_WAVES][32];

    const int tid = threadIdx.x;
    fill_bt(bt1, fw1, R_DIM, KP1, tid, 128);
    fill_bt(bt2, fw2, F_DIM, F_DIM, tid, 128);
    __syncthreads();

    const int lane = tid & 31, wave = tid >> 5;
    const int half = lane >> 4, l15 = lane & 15;
    const float inv_cut = PIF / cutoffp[0];
    __bf16* mysc = sc[wave];
    v8f z = {0.f, 0.f, 0.f, 0.f, 0.f, 0.f, 0.f, 0.f};

#if HAVE_TDM
    const uint32_t lds_b0 = (uint32_t)(uintptr_t)&rbuf[wave][0][0];
    const uint32_t lds_b1 = (uint32_t)(uintptr_t)&rbuf[wave][1][0];
#endif

    for (int tile = blockIdx.x * EDGE_WAVES + wave; tile < ntiles32;
         tile += gridDim.x * EDGE_WAVES) {
        const int e0 = tile * 32;

        // edge metadata + cutoff factor (one edge per lane; same-wave LDS)
        {
            int e = e0 + lane;
            wfac[wave][lane] = 1.f + __cosf(dist[e] * inv_cut);
            ssrc[wave][lane] = (int)edge_index[(size_t)E + e];
            sdst[wave][lane] = (int)edge_index[e];
        }

        // ---- layer 1: [32 x 300] x [300 x 64], K-pipelined through LDS ----
        v8f acc[2][4] = {{z, z, z, z}, {z, z, z, z}};
#if HAVE_TDM
        tdm_load_2d(lds_b0, rbf + (size_t)e0 * R_DIM, R_DIM, 32, 32, 32, R_DIM);
#endif
        for (int kt = 0; kt < KT1; ++kt) {
            const float* rb = rbuf[wave][kt & 1];
#if HAVE_TDM
            int nxt = kt + 1;
            if (nxt < KT1) {
                tdm_load_2d((nxt & 1) ? lds_b1 : lds_b0,
                            rbf + (size_t)e0 * R_DIM + nxt * 32,
                            (uint32_t)(R_DIM - nxt * 32), 32, 32, 32, R_DIM);
                __builtin_amdgcn_s_wait_tensorcnt(1);
            } else {
                __builtin_amdgcn_s_wait_tensorcnt(0);
            }
#else
            {   // fallback: cooperative synchronous copy of the 32x32 tile
                float* wb = rbuf[wave][kt & 1];
                const float* g = rbf + (size_t)e0 * R_DIM + kt * 32;
                for (int k = 0; k < 32; ++k) {
                    int kk = kt * 32 + k;
                    wb[lane * 32 + k] = (kk < R_DIM) ? g[(size_t)lane * R_DIM + k] : 0.f;
                }
            }
#endif
            v16bf a0 = load_a_lds_f32(rb + l15 * 32, half);
            v16bf a1 = load_a_lds_f32(rb + (16 + l15) * 32, half);
#pragma unroll
            for (int nt = 0; nt < 4; ++nt) {
                v16bf bb = load_frag_lds(bt1, KP1, nt * 16 + l15, kt * 32, half);
                acc[0][nt] = wmma_bf16(a0, bb, acc[0][nt]);
                acc[1][nt] = wmma_bf16(a1, bb, acc[1][nt]);
            }
        }

        // bias + ssp, re-layout C->A through LDS as bf16 [m][c]
#pragma unroll
        for (int mt = 0; mt < 2; ++mt)
#pragma unroll
            for (int nt = 0; nt < 4; ++nt) {
                int c = nt * 16 + l15;
                float bias = fb1[c];
#pragma unroll
                for (int r = 0; r < 8; ++r)
                    mysc[(mt * 16 + r + 8 * half) * F_DIM + c] =
                        (__bf16)sspf(acc[mt][nt][r] + bias);
            }

        // ---- layer 2: [32 x 64] x [64 x 64] ----
        v8f acc2[2][4] = {{z, z, z, z}, {z, z, z, z}};
#pragma unroll
        for (int k0 = 0; k0 < F_DIM; k0 += 32) {
            v16bf a0 = load_frag_lds(mysc, F_DIM, l15, k0, half);
            v16bf a1 = load_frag_lds(mysc, F_DIM, 16 + l15, k0, half);
#pragma unroll
            for (int nt = 0; nt < 4; ++nt) {
                v16bf bb = load_frag_lds(bt2, F_DIM, nt * 16 + l15, k0, half);
                acc2[0][nt] = wmma_bf16(a0, bb, acc2[0][nt]);
                acc2[1][nt] = wmma_bf16(a1, bb, acc2[1][nt]);
            }
        }

        // bias + ssp + cutoff, gather h[src], scatter-add into agg[dst]
#pragma unroll
        for (int mt = 0; mt < 2; ++mt)
#pragma unroll
            for (int nt = 0; nt < 4; ++nt) {
                int c = nt * 16 + l15;
                float bias = fb2[c];
#pragma unroll
                for (int r = 0; r < 8; ++r) {
                    int m = mt * 16 + r + 8 * half;
                    float w = sspf(acc2[mt][nt][r] + bias) * wfac[wave][m];
                    float msg = w * h[(size_t)ssrc[wave][m] * F_DIM + c];
                    atomicAdd(&agg[(size_t)sdst[wave][m] * F_DIM + c], msg);
                }
            }
    }
}

// -------- out = ssp(agg@l2w+l2b)@l3w + l3b + x0 --------
__global__ __launch_bounds__(256) void node_out(
    const float* __restrict__ agg, const float* __restrict__ l2w,
    const float* __restrict__ l2b, const float* __restrict__ l3w,
    const float* __restrict__ l3b, const float* __restrict__ x0,
    float* __restrict__ out, int ntiles) {
    __shared__ __bf16 bt2[F_DIM * F_DIM];
    __shared__ __bf16 bt3[F_DIM * F_DIM];
    __shared__ __bf16 sc[8][16 * F_DIM];
    const int tid = threadIdx.x;
    fill_bt(bt2, l2w, F_DIM, F_DIM, tid, 256);
    fill_bt(bt3, l3w, F_DIM, F_DIM, tid, 256);
    __syncthreads();

    const int lane = tid & 31, wave = tid >> 5;
    const int half = lane >> 4, l15 = lane & 15;
    __bf16* mysc = sc[wave];
    v8f z = {0.f, 0.f, 0.f, 0.f, 0.f, 0.f, 0.f, 0.f};

    for (int tile = blockIdx.x * 8 + wave; tile < ntiles; tile += gridDim.x * 8) {
        const int m0 = tile * 16;
        v8f acc[4] = {z, z, z, z};
#pragma unroll
        for (int k0 = 0; k0 < F_DIM; k0 += 32) {
            v16bf a = load_a_global(agg, m0 + l15, F_DIM, k0, F_DIM, half);
#pragma unroll
            for (int nt = 0; nt < 4; ++nt) {
                v16bf bb = load_frag_lds(bt2, F_DIM, nt * 16 + l15, k0, half);
                acc[nt] = wmma_bf16(a, bb, acc[nt]);
            }
        }
#pragma unroll
        for (int nt = 0; nt < 4; ++nt) {
            int c = nt * 16 + l15;
            float bias = l2b[c];
#pragma unroll
            for (int r = 0; r < 8; ++r)
                mysc[(r + 8 * half) * F_DIM + c] = (__bf16)sspf(acc[nt][r] + bias);
        }
        v8f acc2[4] = {z, z, z, z};
#pragma unroll
        for (int k0 = 0; k0 < F_DIM; k0 += 32) {
            v16bf a = load_frag_lds(mysc, F_DIM, l15, k0, half);
#pragma unroll
            for (int nt = 0; nt < 4; ++nt) {
                v16bf bb = load_frag_lds(bt3, F_DIM, nt * 16 + l15, k0, half);
                acc2[nt] = wmma_bf16(a, bb, acc2[nt]);
            }
        }
#pragma unroll
        for (int nt = 0; nt < 4; ++nt) {
            int c = nt * 16 + l15;
            float bias = l3b[c];
#pragma unroll
            for (int r = 0; r < 8; ++r) {
                size_t off = (size_t)(m0 + r + 8 * half) * F_DIM + c;
                out[off] = acc2[nt][r] + bias + x0[off];
            }
        }
    }
}

extern "C" void kernel_launch(void* const* d_in, const int* in_sizes, int n_in,
                              void* d_out, int out_size, void* d_ws, size_t ws_size,
                              hipStream_t stream) {
    const long long* edge_index = (const long long*)d_in[0];   // int64 [2,E]
    const float* node_feature   = (const float*)d_in[1];
    const float* rbf            = (const float*)d_in[2];
    const float* dist           = (const float*)d_in[3];
    const float* cutoff         = (const float*)d_in[4];
    const float* fw1 = (const float*)d_in[5];
    const float* fb1 = (const float*)d_in[6];
    const float* fw2 = (const float*)d_in[7];
    const float* fb2 = (const float*)d_in[8];
    const float* l1w = (const float*)d_in[9];
    const float* l1b = (const float*)d_in[10];
    const float* l2w = (const float*)d_in[11];
    const float* l2b = (const float*)d_in[12];
    const float* l3w = (const float*)d_in[13];
    const float* l3b = (const float*)d_in[14];

    const int E = in_sizes[3];            // dist has E elements
    const int N = in_sizes[1] / F_DIM;    // node_feature is N*64

    float* h   = (float*)d_ws;            // [N][64]
    float* agg = h + (size_t)N * F_DIM;   // [N][64]

    zero_kernel<<<(N * F_DIM + 255) / 256, 256, 0, stream>>>(agg, N * F_DIM);
    node_linear1<<<160, 256, 0, stream>>>(node_feature, l1w, l1b, h, (N + 15) / 16);
    edge_kernel<<<960, 128, 0, stream>>>(edge_index, rbf, dist, cutoff,
                                         fw1, fb1, fw2, fb2, h, agg,
                                         E, (E + 31) / 32);
    node_out<<<160, 256, 0, stream>>>(agg, l2w, l2b, l3w, l3b, node_feature,
                                      (float*)d_out, (N + 15) / 16);
}
